// myNNConv_82798379532675
// MI455X (gfx1250) — compile-verified, matrix operand order checked by reading, not run on hardware
//
#include <hip/hip_runtime.h>
#include <hip/hip_bf16.h>

typedef __attribute__((ext_vector_type(16))) _Float16 v16h;
typedef __attribute__((ext_vector_type(8)))  _Float16 v8h;
typedef __attribute__((ext_vector_type(8)))  float    v8f;

#define N_NODES 10000
#define N_EDGES 160000
#define C 256
#define ENC_NEG_INF 0x007FFFFFu   // encode(bits(-inf)=0xFF800000) = ~bits

__device__ __forceinline__ v8f wmma16(v16h a, v16h b, v8f c) {
    return __builtin_amdgcn_wmma_f32_16x16x32_f16(
        /*neg_a=*/false, a, /*neg_b=*/false, b,
        /*c_mod=*/(short)0, c, /*reuse_a=*/false, /*reuse_b=*/false);
}

// Load a 16x32 f16 fragment (A-layout; B uses same pattern indexed by N).
// rowbase points at this lane's row (row = tileBase + (lane&15)), ld = 256.
// hi = lane>>4. Per-lane: 8 halves at col kk + hi*8, and 8 more at +16.
__device__ __forceinline__ v16h load_frag(const _Float16* rowbase, int kk, int hi) {
    const int c0 = kk + hi * 8;
    v8h lo = *(const v8h*)(rowbase + c0);
    v8h hg = *(const v8h*)(rowbase + c0 + 16);
    v16h r;
#pragma unroll
    for (int i = 0; i < 8; ++i) { r[i] = lo[i]; r[8 + i] = hg[i]; }
    return r;
}

__device__ __forceinline__ unsigned enc_f32(float f) {
    unsigned b = __float_as_uint(f);
    return (b & 0x80000000u) ? ~b : (b | 0x80000000u);
}
__device__ __forceinline__ float dec_f32(unsigned u) {
    if (u == ENC_NEG_INF) return 0.0f;                 // empty segment -> 0
    return (u & 0x80000000u) ? __uint_as_float(u & 0x7FFFFFFFu)
                             : __uint_as_float(~u);
}

// ---------------- prep: weights ----------------
// UT[n][k] = W1[k][n] - W1[k+256][n];  VT[n][k] = W1[k+256][n];  W2T[n][k] = W2[k][n]
__global__ __launch_bounds__(256) void prep_weights(const float* __restrict__ W1,
                                                    const float* __restrict__ W2,
                                                    _Float16* __restrict__ UT,
                                                    _Float16* __restrict__ VT,
                                                    _Float16* __restrict__ W2T) {
    int idx = blockIdx.x * 256 + threadIdx.x;     // 65536 total
    int n = idx & (C - 1);
    int k = idx >> 8;
    float wtop = W1[k * C + n];
    float wbot = W1[(k + C) * C + n];
    UT [n * C + k] = (_Float16)(wtop - wbot);
    VT [n * C + k] = (_Float16)wbot;
    W2T[n * C + k] = (_Float16)W2[k * C + n];
}

// ---------------- prep: x -> f16, init agg ----------------
__global__ __launch_bounds__(256) void prep_x_init(const float* __restrict__ x,
                                                   _Float16* __restrict__ xh,
                                                   unsigned* __restrict__ aggE) {
    int idx = blockIdx.x * 256 + threadIdx.x;     // N_NODES*C total
    xh[idx]   = (_Float16)x[idx];
    aggE[idx] = ENC_NEG_INF;
}

// ---------------- node GEMM: P = x@U + b1, Q = x@V ----------------
// grid = 625 (16-node tiles), block = 128 (4 waves); wave w does N-tiles [4w..4w+3]
__global__ __launch_bounds__(128) void node_gemm(const _Float16* __restrict__ xh,
                                                 const _Float16* __restrict__ UT,
                                                 const _Float16* __restrict__ VT,
                                                 const float* __restrict__ b1,
                                                 float* __restrict__ P,
                                                 float* __restrict__ Q) {
    const int tile = blockIdx.x;
    const int lane = threadIdx.x & 31;
    const int wave = threadIdx.x >> 5;
    const int r    = lane & 15;
    const int hi   = lane >> 4;
    const _Float16* arow = xh + (size_t)(tile * 16 + r) * C;

    v8f accU[4] = {}, accV[4] = {};
#pragma unroll
    for (int kk = 0; kk < C; kk += 32) {
        v16h a = load_frag(arow, kk, hi);
#pragma unroll
        for (int t = 0; t < 4; ++t) {
            const int n0 = (wave * 4 + t) * 16 + r;
            v16h bu = load_frag(UT + (size_t)n0 * C, kk, hi);
            accU[t] = wmma16(a, bu, accU[t]);
            v16h bv = load_frag(VT + (size_t)n0 * C, kk, hi);
            accV[t] = wmma16(a, bv, accV[t]);
        }
    }
#pragma unroll
    for (int t = 0; t < 4; ++t) {
        const int n = (wave * 4 + t) * 16 + r;   // C/D: lane -> column
        const float bb = b1[n];
#pragma unroll
        for (int m = 0; m < 8; ++m) {            // rows hi*8 + m
            const int row = tile * 16 + hi * 8 + m;
            P[(size_t)row * C + n] = accU[t][m] + bb;
            Q[(size_t)row * C + n] = accV[t][m];
        }
    }
}

// ---------------- edge GEMM + segment max ----------------
// grid = 10000 (16-edge tiles), block = 256 (8 waves); wave w does N-tiles {w, w+8}
__global__ __launch_bounds__(256) void edge_gemm(const float* __restrict__ P,
                                                 const float* __restrict__ Q,
                                                 const _Float16* __restrict__ W2T,
                                                 const float* __restrict__ b2,
                                                 const int* __restrict__ ei,
                                                 unsigned* __restrict__ aggE) {
    __shared__ __align__(16) _Float16 sh1[16 * C];   // h1 tile: 16 edges x 256
    __shared__ int ssrc[16], sdst[16];

    const int tile = blockIdx.x;
    const int t = threadIdx.x;
    if (t < 16) {
        ssrc[t] = ei[tile * 16 + t];
        sdst[t] = ei[N_EDGES + tile * 16 + t];
    }
    __syncthreads();

    // stage h1 = relu(P[dst] + Q[src]) into LDS (b1 already folded into P)
    const int c = t;                                  // column 0..255
#pragma unroll 4
    for (int e = 0; e < 16; ++e) {
        float v = P[(size_t)sdst[e] * C + c] + Q[(size_t)ssrc[e] * C + c];
        v = v > 0.0f ? v : 0.0f;
        sh1[e * C + c] = (_Float16)v;
    }
    __syncthreads();

    const int lane = t & 31, wave = t >> 5;
    const int r = lane & 15, hi = lane >> 4;
    const int nt[2] = { wave, wave + 8 };

    v8f acc[2] = {};
#pragma unroll
    for (int kk = 0; kk < C; kk += 32) {
        v16h a = load_frag(sh1 + r * C, kk, hi);      // ds_load_b128 x2
#pragma unroll
        for (int j = 0; j < 2; ++j) {
            v16h b = load_frag(W2T + (size_t)(nt[j] * 16 + r) * C, kk, hi);
            acc[j] = wmma16(a, b, acc[j]);
        }
    }
#pragma unroll
    for (int j = 0; j < 2; ++j) {
        const int n = nt[j] * 16 + r;
        const float bb = b2[n];
#pragma unroll
        for (int m = 0; m < 8; ++m) {
            const int e = hi * 8 + m;                 // edge row within tile
            const float v = acc[j][m] + bb;
            atomicMax(&aggE[(size_t)sdst[e] * C + n], enc_f32(v));
        }
    }
}

// ---------------- BatchNorm stats: one block per column ----------------
__global__ __launch_bounds__(256) void bn_stats(const unsigned* __restrict__ aggE,
                                                float* __restrict__ mean,
                                                float* __restrict__ rstd) {
    __shared__ float ss[256], sq[256];
    const int col = blockIdx.x;
    float s = 0.0f, q = 0.0f;
    for (int i = threadIdx.x; i < N_NODES; i += 256) {
        float v = dec_f32(aggE[(size_t)i * C + col]);
        s += v; q += v * v;
    }
    ss[threadIdx.x] = s; sq[threadIdx.x] = q;
    __syncthreads();
    for (int off = 128; off > 0; off >>= 1) {
        if (threadIdx.x < off) {
            ss[threadIdx.x] += ss[threadIdx.x + off];
            sq[threadIdx.x] += sq[threadIdx.x + off];
        }
        __syncthreads();
    }
    if (threadIdx.x == 0) {
        float m = ss[0] * (1.0f / N_NODES);
        float var = sq[0] * (1.0f / N_NODES) - m * m;
        mean[col] = m;
        rstd[col] = rsqrtf(var + 1e-5f);
    }
}

// ---------------- BatchNorm apply ----------------
__global__ __launch_bounds__(256) void bn_apply(const unsigned* __restrict__ aggE,
                                                const float* __restrict__ mean,
                                                const float* __restrict__ rstd,
                                                const float* __restrict__ gamma,
                                                const float* __restrict__ beta,
                                                float* __restrict__ out) {
    int idx = blockIdx.x * 256 + threadIdx.x;         // N_NODES*C
    int col = threadIdx.x;                            // blockDim.x == C
    float v = dec_f32(aggE[idx]);
    out[idx] = (v - mean[col]) * rstd[col] * gamma[col] + beta[col];
}

extern "C" void kernel_launch(void* const* d_in, const int* in_sizes, int n_in,
                              void* d_out, int out_size, void* d_ws, size_t ws_size,
                              hipStream_t stream) {
    const float* x   = (const float*)d_in[0];
    const int*   ei  = (const int*)  d_in[1];
    const float* W1  = (const float*)d_in[2];
    const float* b1  = (const float*)d_in[3];
    const float* W2  = (const float*)d_in[4];
    const float* b2  = (const float*)d_in[5];
    const float* gmm = (const float*)d_in[6];
    const float* bta = (const float*)d_in[7];
    float* out = (float*)d_out;

    char* ws = (char*)d_ws;
    size_t off = 0;
    _Float16* xh  = (_Float16*)(ws + off); off += (size_t)N_NODES * C * 2;   // 5,120,000
    _Float16* UT  = (_Float16*)(ws + off); off += (size_t)C * C * 2;         // 131,072
    _Float16* VT  = (_Float16*)(ws + off); off += (size_t)C * C * 2;
    _Float16* W2T = (_Float16*)(ws + off); off += (size_t)C * C * 2;
    float*    P   = (float*)   (ws + off); off += (size_t)N_NODES * C * 4;   // 10,240,000
    float*    Q   = (float*)   (ws + off); off += (size_t)N_NODES * C * 4;
    unsigned* agg = (unsigned*)(ws + off); off += (size_t)N_NODES * C * 4;
    float*    mu  = (float*)   (ws + off); off += 1024;
    float*    rs  = (float*)   (ws + off); off += 1024;                      // ~36.2 MB total

    prep_weights<<<(C * C) / 256, 256, 0, stream>>>(W1, W2, UT, VT, W2T);
    prep_x_init <<<(N_NODES * C) / 256, 256, 0, stream>>>(x, xh, agg);
    node_gemm   <<<N_NODES / 16, 128, 0, stream>>>(xh, UT, VT, b1, P, Q);
    edge_gemm   <<<N_EDGES / 16, 256, 0, stream>>>(P, Q, W2T, b2, ei, agg);
    bn_stats    <<<C, 256, 0, stream>>>(agg, mu, rs);
    bn_apply    <<<(N_NODES * C) / 256, 256, 0, stream>>>(agg, mu, rs, gmm, bta, out);
}
